// ResidualReflectance_72713796321298
// MI455X (gfx1250) — matrix operand order repaired and checked by reading, not compiled, not run on hardware
//
#include <hip/hip_runtime.h>

typedef _Float16 v16h __attribute__((ext_vector_type(16)));
typedef _Float16 v8h  __attribute__((ext_vector_type(8)));
typedef float    v8f  __attribute__((ext_vector_type(8)));

// f16 workspace layout (element offsets)
#define OFF0 0                      // W0: 512 x 128 (K padded 100->128)
#define OFF1 (512*128)              // W1: 512 x 512
#define OFF2 (OFF1 + 512*512)       // W2: 512 x 512
#define OFF3 (OFF2 + 512*512)       // W3: 512 x 512
#define OFF4 (OFF3 + 512*512)       // W4: 16 x 512 (rows 3..15 zero)
#define OFFL (OFF4 + 16*512)        // WL: 384 x 128 (K padded 100->128)

#define ENC_STRIDE 136              // 64-bank-rotating LDS strides (multiple of 8 halves)
#define H_STRIDE   520

__device__ __forceinline__ v16h cat16(v8h lo, v8h hi) {
  v16h r;
#pragma unroll
  for (int i = 0; i < 8; ++i) { r[i] = lo[i]; r[i + 8] = hi[i]; }
  return r;
}

// A fragment (16x32 f16, MxK): lane holds M=l16; K chunks at hi*8 and 16+hi*8
__device__ __forceinline__ v16h load_a_frag(const _Float16* rowp, int kb, int hi) {
  const _Float16* p = rowp + kb;
  v8h lo = *(const v8h*)(p + hi * 8);
  v8h hh = *(const v8h*)(p + 16 + hi * 8);
  return cat16(lo, hh);
}

// B fragment (32x16 f16, KxN): lane holds column N=l16; K chunks at hi*8 and 16+hi*8.
// B[k][n] = W[n][k] (row-major W), so these are two 16-byte chunks of row n of W.
__device__ __forceinline__ v16h load_b_frag(const _Float16* p) {
  v8h lo = *(const v8h*)p;
  v8h hh = *(const v8h*)(p + 16);
  return cat16(lo, hh);
}

__device__ __forceinline__ v8f wmma16(v16h a, v16h b, v8f c) {
  return __builtin_amdgcn_wmma_f32_16x16x32_f16(false, a, false, b, (short)0, c,
                                                false, false);
}

// One dense layer: 64 rows (LDS, stride SIN) x W(512 x K, global f16) -> 64x512
// (LDS, stride H_STRIDE), bias + ReLU. 8 waves: wave w -> M-tile pair (w&1),
// N-quarter (w>>1). B fragments reused across 2 M-tiles; A reused across 4 N-tiles.
template <int K, int SIN>
__device__ __forceinline__ void dense_layer(const _Float16* hin, _Float16* hout,
                                            const _Float16* __restrict__ W,
                                            const float* __restrict__ bias,
                                            int wave, int l16, int hi) {
  const int mp = (wave & 1) * 2;      // first of two M-tiles
  const int nq = wave >> 1;           // N-quarter: 8 N-tiles
  const _Float16* arow0 = hin + (mp * 16 + l16) * SIN;
  const _Float16* arow1 = arow0 + 16 * SIN;
#pragma unroll 1
  for (int s = 0; s < 2; ++s) {
    const int n0 = (nq * 8 + s * 4) * 16;   // starting output column of strip
    v8f acc0[4] = {};
    v8f acc1[4] = {};
#pragma unroll 2
    for (int kb = 0; kb < K; kb += 32) {
      v16h a0 = load_a_frag(arow0, kb, hi);
      v16h a1 = load_a_frag(arow1, kb, hi);
#pragma unroll
      for (int t = 0; t < 4; ++t) {
        v16h b = load_b_frag(W + (n0 + t * 16 + l16) * K + kb + hi * 8);
        acc0[t] = wmma16(a0, b, acc0[t]);
        acc1[t] = wmma16(a1, b, acc1[t]);
      }
    }
#pragma unroll
    for (int t = 0; t < 4; ++t) {
      const int n = n0 + t * 16 + l16;
      const float bv = bias[n];
#pragma unroll
      for (int r = 0; r < 8; ++r) {
        const int m0 = mp * 16 + 8 * hi + r;   // C/D layout: M = r + 8*hi
        float x0 = fmaxf(acc0[t][r] + bv, 0.0f);
        float x1 = fmaxf(acc1[t][r] + bv, 0.0f);
        hout[m0 * H_STRIDE + n] = (_Float16)x0;
        hout[(m0 + 16) * H_STRIDE + n] = (_Float16)x1;
      }
    }
  }
}

__global__ __launch_bounds__(256) void rr_prep(
    const float* __restrict__ Wlin,
    const float* __restrict__ v0, const float* __restrict__ g0,
    const float* __restrict__ v1, const float* __restrict__ g1,
    const float* __restrict__ v2, const float* __restrict__ g2,
    const float* __restrict__ v3, const float* __restrict__ g3,
    const float* __restrict__ v4, const float* __restrict__ g4,
    _Float16* __restrict__ ws) {
  const int wid = blockIdx.x * 8 + (threadIdx.x >> 5);   // one wave per row
  const int lane = threadIdx.x & 31;
  if (wid >= 2448) return;

  const float* src = nullptr;
  float g = 1.0f;
  int Kin = 512, Kpad = 512;
  _Float16* dst;
  bool doNorm = true, zero = false;

  if (wid < 512)        { src = v0 + wid * 100; g = g0[wid]; Kin = 100; Kpad = 128; dst = ws + OFF0 + wid * 128; }
  else if (wid < 1024)  { int r = wid - 512;  src = v1 + r * 512; g = g1[r]; dst = ws + OFF1 + r * 512; }
  else if (wid < 1536)  { int r = wid - 1024; src = v2 + r * 512; g = g2[r]; dst = ws + OFF2 + r * 512; }
  else if (wid < 2048)  { int r = wid - 1536; src = v3 + r * 512; g = g3[r]; dst = ws + OFF3 + r * 512; }
  else if (wid < 2064)  { int r = wid - 2048; dst = ws + OFF4 + r * 512;
                          if (r < 3) { src = v4 + r * 512; g = g4[r]; } else zero = true; }
  else                  { int r = wid - 2064; src = Wlin + r * 100; Kin = 100; Kpad = 128;
                          dst = ws + OFFL + r * 128; doNorm = false; }

  float scale = 0.0f;
  if (!zero) {
    if (doNorm) {
      float s = 0.0f;
      for (int k = lane; k < Kin; k += 32) { float t = src[k]; s += t * t; }
#pragma unroll
      for (int o = 16; o >= 1; o >>= 1) s += __shfl_xor(s, o, 32);
      scale = g * rsqrtf(s);
    } else {
      scale = 1.0f;
    }
  }
  for (int k = lane; k < Kpad; k += 32)
    dst[k] = (_Float16)((!zero && k < Kin) ? src[k] * scale : 0.0f);
}

// 148KB LDS -> 2 workgroups (16 waves) per WGP; min-waves-per-eu=4 caps the
// allocator at 256 VGPRs so the inner loop stays free of s_set_vgpr_msb churn.
__global__ __launch_bounds__(256, 4) void rr_fused(
    const float* __restrict__ X, const int* __restrict__ cids,
    const _Float16* __restrict__ ws,
    const float* __restrict__ b0, const float* __restrict__ b1,
    const float* __restrict__ b2, const float* __restrict__ b3,
    const float* __restrict__ b4, float* __restrict__ out, int rows) {
  __shared__ _Float16 sEnc[64 * ENC_STRIDE];
  __shared__ _Float16 sA[64 * H_STRIDE];
  __shared__ _Float16 sB[64 * H_STRIDE];
  __shared__ float sRgb[64 * 3];

  const int tid = threadIdx.x;
  const int wave = tid >> 5;
  const int lane = tid & 31;
  const int l16 = lane & 15;
  const int hi = lane >> 4;
  const int rowBase = blockIdx.x * 64;
  if (rowBase >= rows) return;

  const _Float16* W0 = ws + OFF0;
  const _Float16* W1 = ws + OFF1;
  const _Float16* W2 = ws + OFF2;
  const _Float16* W3 = ws + OFF3;
  const _Float16* W4 = ws + OFF4;
  const _Float16* WL = ws + OFFL;

  // ---- positional encoding into LDS (f16, K padded to 128) ----
  for (int idx = tid; idx < 64 * 5; idx += 256) {
    int r = idx / 5, d = idx - r * 5;
    float x = X[(rowBase + r) * 5 + d];
    _Float16* e = sEnc + r * ENC_STRIDE + d * 20;
#pragma unroll
    for (int j = 0; j < 10; ++j) {
      float xf = x * (float)(1u << j);
      e[2 * j] = (_Float16)__sinf(xf);
      e[2 * j + 1] = (_Float16)__cosf(xf);
    }
  }
  for (int idx = tid; idx < 64 * 28; idx += 256) {
    int r = idx / 28, c = idx - r * 28;
    sEnc[r * ENC_STRIDE + 100 + c] = (_Float16)0.0f;
  }
  __syncthreads();

  // ---- reflectance: per-row gather of 3 W_lin rows, dot with enc ----
  // Both operands are zero-padded past k=100, so run 13 aligned v8h chunks.
  for (int idx = tid; idx < 64 * 3; idx += 256) {
    int r = idx / 3, c = idx - r * 3;
    int cid = cids[rowBase + r];
    const _Float16* wl = WL + (3 * cid + c) * 128;
    const _Float16* e = sEnc + r * ENC_STRIDE;
    float acc = 0.0f;
#pragma unroll
    for (int kc = 0; kc < 104; kc += 8) {
      v8h ev = *(const v8h*)(e + kc);
      v8h wv = *(const v8h*)(wl + kc);
#pragma unroll
      for (int i = 0; i < 8; ++i) acc += (float)ev[i] * (float)wv[i];
    }
    sRgb[idx] = acc;
  }

  // ---- MLP: 100->512->512->512->512 (ReLU) ----
  dense_layer<128, ENC_STRIDE>(sEnc, sA, W0, b0, wave, l16, hi);
  __syncthreads();
  dense_layer<512, H_STRIDE>(sA, sB, W1, b1, wave, l16, hi);
  __syncthreads();
  dense_layer<512, H_STRIDE>(sB, sA, W2, b2, wave, l16, hi);
  __syncthreads();
  dense_layer<512, H_STRIDE>(sA, sB, W3, b3, wave, l16, hi);
  __syncthreads();

  // ---- final 512->3 (padded to 16), tanh, + rgb ----
  if (wave < 4) {
    const int mt = wave;
    const _Float16* arow = sB + (mt * 16 + l16) * H_STRIDE;
    v8f acc = {};
    for (int kb = 0; kb < 512; kb += 32) {
      v16h a = load_a_frag(arow, kb, hi);
      v16h b = load_b_frag(W4 + l16 * 512 + kb + hi * 8);
      acc = wmma16(a, b, acc);
    }
    if (l16 < 3) {
      const float bv = b4[l16];
#pragma unroll
      for (int r = 0; r < 8; ++r) {
        const int m = mt * 16 + 8 * hi + r;
        float v = tanhf(acc[r] + bv) + sRgb[m * 3 + l16];
        out[(rowBase + m) * 3 + l16] = v;
      }
    }
  }
}

extern "C" void kernel_launch(void* const* d_in, const int* in_sizes, int n_in,
                              void* d_out, int out_size, void* d_ws,
                              size_t ws_size, hipStream_t stream) {
  const float* X    = (const float*)d_in[0];
  const int*   cids = (const int*)d_in[1];
  const float* Wlin = (const float*)d_in[2];
  const float* v0 = (const float*)d_in[3];  const float* g0 = (const float*)d_in[4];  const float* b0 = (const float*)d_in[5];
  const float* v1 = (const float*)d_in[6];  const float* g1 = (const float*)d_in[7];  const float* b1 = (const float*)d_in[8];
  const float* v2 = (const float*)d_in[9];  const float* g2 = (const float*)d_in[10]; const float* b2 = (const float*)d_in[11];
  const float* v3 = (const float*)d_in[12]; const float* g3 = (const float*)d_in[13]; const float* b3 = (const float*)d_in[14];
  const float* v4 = (const float*)d_in[15]; const float* g4 = (const float*)d_in[16]; const float* b4 = (const float*)d_in[17];

  _Float16* ws = (_Float16*)d_ws;
  float* out = (float*)d_out;
  const int rows = in_sizes[0] / 5;

  rr_prep<<<306, 256, 0, stream>>>(Wlin, v0, g0, v1, g1, v2, g2, v3, g3, v4, g4, ws);
  rr_fused<<<rows / 64, 256, 0, stream>>>(X, cids, ws, b0, b1, b2, b3, b4, out, rows);
}
